// ExampleModel_9234179686517
// MI455X (gfx1250) — matrix operand order repaired and checked by reading, not compiled, not run on hardware
//
#include <hip/hip_runtime.h>

typedef __attribute__((ext_vector_type(16))) _Float16 v16h;
typedef __attribute__((ext_vector_type(8)))  _Float16 v8h;
typedef __attribute__((ext_vector_type(8)))  float    v8f;

#define HID 16
#define FMH 512
#define FMW 512

__global__ __launch_bounds__(256) void nerf_grid_mlp_wmma(
    const float* __restrict__ xin,
    const float* __restrict__ fm,
    const float* __restrict__ W1, const float* __restrict__ b1,
    const float* __restrict__ W2, const float* __restrict__ b2,
    const float* __restrict__ W3, const float* __restrict__ b3,
    const float* __restrict__ W4, const float* __restrict__ b4,
    const float* __restrict__ W5, const float* __restrict__ b5,
    float* __restrict__ out, int npts)
{
    // Per-wave private LDS: 32-row (2 tiles x 16) x 32-col f16 activation tile
    // (cols >= 16, and 7..15 for layer 1, stay zero = K padding), plus a
    // 96-float staging buffer for coalesced global IO.
    __shared__ _Float16 act[8][32][32];   // 16 KB
    __shared__ float    stage[8][96];     // 3 KB

    const int w    = threadIdx.x >> 5;
    const int lane = threadIdx.x & 31;
    const int col  = lane & 15;    // B/C/D column owned by this lane
    const int hi   = lane >> 4;    // lane half selector

    const long long base  = ((long long)blockIdx.x * 8 + w) * 32; // first point of wave
    const long long base3 = base * 3;
    const long long n3    = (long long)npts * 3;

    // ---- 1) zero the activation tile once (padding stays zero all layers) ----
    {
        float4 z4 = make_float4(0.f, 0.f, 0.f, 0.f);
        float4* rp = (float4*)(&act[w][lane][0]);   // 64B row per lane
        rp[0] = z4; rp[1] = z4; rp[2] = z4; rp[3] = z4;
    }

    // ---- 2) coalesced load of 32 points (96 contiguous floats) ----
    #pragma unroll
    for (int k = 0; k < 3; ++k) {
        long long gi = base3 + lane + k * 32;
        stage[w][lane + k * 32] = xin[gi < n3 ? gi : (n3 - 1)];
    }
    asm volatile("" ::: "memory");   // wave-internal DS ordering is in-order in HW

    // ---- 3) bilinear featuremap sample, write 7-feature f16 row ----
    {
        float cx = stage[w][lane * 3 + 0];
        float cy = stage[w][lane * 3 + 1];
        float cz = stage[w][lane * 3 + 2];
        // reference: fx = clip(u*W - 0.5, 0, W-1)
        float fx = fminf(fmaxf(cx * (float)FMW - 0.5f, 0.f), (float)(FMW - 1));
        float fy = fminf(fmaxf(cy * (float)FMH - 0.5f, 0.f), (float)(FMH - 1));
        float x0f = floorf(fx), y0f = floorf(fy);
        float wx = fx - x0f,   wy = fy - y0f;
        int ix0 = (int)x0f, iy0 = (int)y0f;
        int ix1 = min(ix0 + 1, FMW - 1), iy1 = min(iy0 + 1, FMH - 1);
        float w00 = (1.f - wy) * (1.f - wx), w01 = (1.f - wy) * wx;
        float w10 = wy * (1.f - wx),         w11 = wy * wx;
        int o00 = iy0 * FMW + ix0, o01 = iy0 * FMW + ix1;
        int o10 = iy1 * FMW + ix0, o11 = iy1 * FMW + ix1;

        act[w][lane][0] = (_Float16)cx;
        act[w][lane][1] = (_Float16)cy;
        act[w][lane][2] = (_Float16)cz;
        #pragma unroll
        for (int c = 0; c < 4; ++c) {
            const float* fmc = fm + (size_t)c * (FMH * FMW);
            float s = fmc[o00] * w00 + fmc[o01] * w01 + fmc[o10] * w10 + fmc[o11] * w11;
            act[w][lane][3 + c] = (_Float16)s;
        }
    }
    asm volatile("" ::: "memory");

    // A fragment, 16x32 f16: lane holds row (lane&15), Ks {kb..kb+15}
    auto loadA = [&](int tile) -> v16h {
        int r  = tile * 16 + col;
        int kb = hi * 16;
        v8h lo = *(const v8h*)(&act[w][r][kb]);
        v8h up = *(const v8h*)(&act[w][r][kb + 8]);
        v16h a;
        #pragma unroll
        for (int e = 0; e < 8; ++e) { a[e] = lo[e]; a[8 + e] = up[e]; }
        return a;
    };
    // D (16x16 f32): lane col = lane&15, elem i -> row i + 8*hi. ReLU + f16 back to LDS.
    auto storeH = [&](const v8f& c, int tile) {
        #pragma unroll
        for (int i = 0; i < 8; ++i)
            act[w][tile * 16 + i + 8 * hi][col] = (_Float16)fmaxf(c[i], 0.f);
    };

    // ---- layer 1: 7 -> 16 (K padded 7 -> 32) ----
    {
        v16h bw;
        #pragma unroll
        for (int e = 0; e < 16; ++e)
            bw[e] = (e < 7 && hi == 0) ? (_Float16)W1[e * HID + col] : (_Float16)0.f;
        float bias = b1[col];
        v8f ci;
        #pragma unroll
        for (int i = 0; i < 8; ++i) ci[i] = bias;
        v16h a0 = loadA(0), a1 = loadA(1);
        v8f d0 = __builtin_amdgcn_wmma_f32_16x16x32_f16(false, a0, false, bw, (short)0, ci, false, false);
        v8f d1 = __builtin_amdgcn_wmma_f32_16x16x32_f16(false, a1, false, bw, (short)0, ci, false, false);
        asm volatile("" ::: "memory");
        storeH(d0, 0); storeH(d1, 1);
    }

    // ---- layers 2..4: 16 -> 16 (K padded 16 -> 32) ----
    const float* Ws[3] = { W2, W3, W4 };
    const float* bs[3] = { b2, b3, b4 };
    #pragma unroll
    for (int L = 0; L < 3; ++L) {
        asm volatile("" ::: "memory");
        v16h bw;
        #pragma unroll
        for (int e = 0; e < 16; ++e)
            bw[e] = (hi == 0) ? (_Float16)Ws[L][e * HID + col] : (_Float16)0.f;
        float bias = bs[L][col];
        v8f ci;
        #pragma unroll
        for (int i = 0; i < 8; ++i) ci[i] = bias;
        v16h a0 = loadA(0), a1 = loadA(1);
        v8f d0 = __builtin_amdgcn_wmma_f32_16x16x32_f16(false, a0, false, bw, (short)0, ci, false, false);
        v8f d1 = __builtin_amdgcn_wmma_f32_16x16x32_f16(false, a1, false, bw, (short)0, ci, false, false);
        asm volatile("" ::: "memory");
        storeH(d0, 0); storeH(d1, 1);
    }

    // ---- layer 5: 16 -> 3 (N padded 3 -> 16), no ReLU ----
    {
        asm volatile("" ::: "memory");
        v16h bw;
        #pragma unroll
        for (int e = 0; e < 16; ++e)
            bw[e] = (hi == 0 && col < 3) ? (_Float16)W5[e * 3 + col] : (_Float16)0.f;
        float bias = (col < 3) ? b5[col] : 0.f;
        v8f ci;
        #pragma unroll
        for (int i = 0; i < 8; ++i) ci[i] = bias;
        v16h a0 = loadA(0), a1 = loadA(1);
        v8f d0 = __builtin_amdgcn_wmma_f32_16x16x32_f16(false, a0, false, bw, (short)0, ci, false, false);
        v8f d1 = __builtin_amdgcn_wmma_f32_16x16x32_f16(false, a1, false, bw, (short)0, ci, false, false);

        // stage f32 outputs so the global store is 96 contiguous floats per wave
        if (col < 3) {
            #pragma unroll
            for (int i = 0; i < 8; ++i) {
                stage[w][(i + 8 * hi) * 3 + col]       = d0[i];
                stage[w][(16 + i + 8 * hi) * 3 + col]  = d1[i];
            }
        }
        asm volatile("" ::: "memory");
        #pragma unroll
        for (int k = 0; k < 3; ++k) {
            long long gi = base3 + lane + k * 32;
            if (gi < n3) out[gi] = stage[w][lane + k * 32];
        }
    }
}

extern "C" void kernel_launch(void* const* d_in, const int* in_sizes, int n_in,
                              void* d_out, int out_size, void* d_ws, size_t ws_size,
                              hipStream_t stream) {
    const float* xin = (const float*)d_in[0];
    const float* fm  = (const float*)d_in[1];
    const float* W1  = (const float*)d_in[2];
    const float* b1  = (const float*)d_in[3];
    const float* W2  = (const float*)d_in[4];
    const float* b2  = (const float*)d_in[5];
    const float* W3  = (const float*)d_in[6];
    const float* b3  = (const float*)d_in[7];
    const float* W4  = (const float*)d_in[8];
    const float* b4  = (const float*)d_in[9];
    const float* W5  = (const float*)d_in[10];
    const float* b5  = (const float*)d_in[11];
    float* out = (float*)d_out;

    int npts = in_sizes[0] / 3;                 // N points (x is N x 3)
    int nblocks = (npts + 255) / 256;           // 256 threads = 8 waves * 32 points
    nerf_grid_mlp_wmma<<<nblocks, 256, 0, stream>>>(
        xin, fm, W1, b1, W2, b2, W3, b3, W4, b4, W5, b5, out, npts);
}
